// SimpleMultiHeadAttention_55551107006598
// MI455X (gfx1250) — matrix operand order repaired and checked by reading, not compiled, not run on hardware
//
#include <hip/hip_runtime.h>
#include <cstdint>
#include <cstddef>

#define D_MODEL 1024
#define NHEAD   16
#define D_HEAD  64
#define BATCH   2
#define SEQ     2048
#define MROWS   (BATCH * SEQ)   // 4096

#define LDS_PAD 8               // +16B per row: preserves b128 alignment,
                                // rotates bank mapping row-to-row

typedef __attribute__((ext_vector_type(16))) __bf16 v16bf;
typedef __attribute__((ext_vector_type(8)))  float  v8f;

union FragU {
    v16bf v;
    uint4 q[2];
};

__device__ __forceinline__ unsigned short f32_to_bf16(float f) {
    union { __bf16 h; unsigned short u; } c;
    c.h = (__bf16)f;                   // native cvt on gfx1250
    return c.u;
}

__device__ __forceinline__ unsigned short ld_bf16(const float* p)          { return f32_to_bf16(*p); }
__device__ __forceinline__ unsigned short ld_bf16(const unsigned short* p) { return *p; }

// Async global->LDS copy (CDNA5): 16B per lane, ASYNCcnt-tracked, no VGPR data.
__device__ __forceinline__ void async_copy_b128(unsigned lds_off, const void* gaddr) {
    asm volatile("global_load_async_to_lds_b128 %0, %1, off"
                 :: "v"(lds_off), "v"(gaddr) : "memory");
}
__device__ __forceinline__ void wait_async0() {
    asm volatile("s_wait_asynccnt 0x0" ::: "memory");
}

// ---------------------------------------------------------------------------
// C = (X(MxK) @ W(KxN) + bias) * outScale.  bf16 WMMA, fp32 accumulate.
// Block tile 128x128, BK=64, 8 waves (2x4), each wave 64x32 (4x2 WMMA accs),
// 16 WMMAs per barrier epoch.
// OUT_MODE 0: bf16 out, head-split layout [B][H][SEQ][D_HEAD]  (QKV proj)
// OUT_MODE 2: fp32 out, row-major [M][N]                       (final proj)
// ---------------------------------------------------------------------------
template<typename XT, int OUT_MODE>
__global__ __launch_bounds__(256)
void gemm_wmma_kernel(const XT* __restrict__ X, const float* __restrict__ W,
                      const float* __restrict__ bias, void* __restrict__ Y,
                      int M, int N, int K, float outScale)
{
    __shared__ unsigned short As[128][64 + LDS_PAD];   // bf16 raw, [m][k]
    __shared__ unsigned short Bs[128][64 + LDS_PAD];   // bf16 raw, [n][k] (W^T)

    const int tid  = threadIdx.x;
    const int wave = tid >> 5;
    const int lane = tid & 31;
    const int half = lane >> 4;     // 0/1
    const int lm   = lane & 15;
    const int wr   = wave >> 2;     // 0..1
    const int wc   = wave & 3;      // 0..3
    const int m0   = blockIdx.y * 128;
    const int n0   = blockIdx.x * 128;

    v8f acc[4][2];
    #pragma unroll
    for (int ms = 0; ms < 4; ++ms)
        #pragma unroll
        for (int ns = 0; ns < 2; ++ns)
            #pragma unroll
            for (int i = 0; i < 8; ++i) acc[ms][ns][i] = 0.f;

    const int arow  = tid >> 1;           // 0..127
    const int apart = (tid & 1) * 32;     // 0 or 32
    const int bk    = tid & 63;           // k row within tile
    const int bseg  = (tid >> 6) * 32;    // n segment base (0,32,64,96)

    for (int kt = 0; kt < K; kt += 64) {
        // ---- stage A tile (convert to bf16) ----
        {
            const XT* xp = X + (size_t)(m0 + arow) * K + kt + apart;
            #pragma unroll
            for (int i = 0; i < 32; ++i) As[arow][apart + i] = ld_bf16(xp + i);
        }
        // ---- stage W tile transposed: Bs[n][k] ----
        {
            const float* wp = W + (size_t)(kt + bk) * N + n0 + bseg;
            #pragma unroll
            for (int i = 0; i < 32; ++i) Bs[bseg + i][bk] = f32_to_bf16(wp[i]);
        }
        __syncthreads();

        // ---- fragments (ISA 16-bit A 16x32 / B 32x16 layouts) ----
        #pragma unroll
        for (int kc = 0; kc < 2; ++kc) {
            FragU a[4], b[2];
            #pragma unroll
            for (int ms = 0; ms < 4; ++ms) {
                const unsigned short* ap = &As[wr * 64 + ms * 16 + lm][kc * 32];
                a[ms].q[0] = *(const uint4*)(ap + 8 * half);        // K = 8h..8h+7
                a[ms].q[1] = *(const uint4*)(ap + 16 + 8 * half);   // K = 16+8h..+7
            }
            #pragma unroll
            for (int ns = 0; ns < 2; ++ns) {
                const unsigned short* bp = &Bs[wc * 32 + ns * 16 + lm][kc * 32];
                b[ns].q[0] = *(const uint4*)(bp + 8 * half);        // K = 8h..8h+7
                b[ns].q[1] = *(const uint4*)(bp + 16 + 8 * half);   // K = 16+8h..+7
            }
            #pragma unroll
            for (int ms = 0; ms < 4; ++ms)
                #pragma unroll
                for (int ns = 0; ns < 2; ++ns)
                    acc[ms][ns] = __builtin_amdgcn_wmma_f32_16x16x32_bf16(
                        false, a[ms].v, false, b[ns].v, (short)0, acc[ms][ns],
                        false, false);
        }
        __syncthreads();
    }

    // ---- epilogue: C layout vgpr r -> M = r + 8*half, lane -> N ----
    #pragma unroll
    for (int ms = 0; ms < 4; ++ms)
        #pragma unroll
        for (int ns = 0; ns < 2; ++ns)
            #pragma unroll
            for (int r = 0; r < 8; ++r) {
                const int gm = m0 + wr * 64 + ms * 16 + r + 8 * half;
                const int gc = n0 + wc * 32 + ns * 16 + lm;
                const float val = (acc[ms][ns][r] + bias[gc]) * outScale;
                if constexpr (OUT_MODE == 0) {
                    const int bidx = gm >> 11, n = gm & 2047;
                    const int h = gc >> 6,    d = gc & 63;
                    ((unsigned short*)Y)[(((size_t)(bidx * NHEAD + h)) * SEQ + n) * D_HEAD + d]
                        = f32_to_bf16(val);
                } else {
                    ((float*)Y)[(size_t)gm * N + gc] = val;
                }
            }
}

// ---------------------------------------------------------------------------
// Flash attention over bf16 Q/K/V in [B*H][SEQ][64] layout.
// Q is pre-scaled by 1/sqrt(d_head) at projection time.
// Block = 8 waves; each wave owns 16 q-rows (q-tile 128); key block = 128.
// K block staged to LDS once per block via async global->LDS copies
// (shared by all 8 waves); V staged transposed; online softmax in fp32.
// 32 WMMAs per key-block iteration per wave.
// ---------------------------------------------------------------------------
__global__ __launch_bounds__(256)
void flash_attn_kernel(const unsigned short* __restrict__ Q,
                       const unsigned short* __restrict__ K,
                       const unsigned short* __restrict__ V,
                       unsigned short* __restrict__ Ctx)
{
    __shared__ unsigned short Ks[128][64 + LDS_PAD];    // K block [key][d]
    __shared__ unsigned short Vt[64][128 + LDS_PAD];    // V^T     [d][key]
    __shared__ unsigned short Pbuf[8][16][128 + LDS_PAD]; // per-wave P tile

    const int tid  = threadIdx.x;
    const int wave = tid >> 5;
    const int lane = tid & 31;
    const int half = lane >> 4;
    const int lm   = lane & 15;
    const int bh   = blockIdx.y;       // batch*head
    const int qt   = blockIdx.x;       // q tile of 128 rows
    const size_t bhoff = (size_t)bh * SEQ * D_HEAD;

    // Q A-fragments, resident for whole kernel (d_head=64 -> 2 K-chunks)
    FragU qf[2];
    {
        const unsigned short* qr =
            Q + bhoff + (size_t)(qt * 128 + wave * 16 + lm) * D_HEAD;
        #pragma unroll
        for (int kc = 0; kc < 2; ++kc) {
            qf[kc].q[0] = *(const uint4*)(qr + kc * 32 + 8 * half);
            qf[kc].q[1] = *(const uint4*)(qr + kc * 32 + 16 + 8 * half);
        }
    }

    float mstate[8], lstate[8];
    #pragma unroll
    for (int r = 0; r < 8; ++r) { mstate[r] = -1e30f; lstate[r] = 0.f; }
    v8f o[4];
    #pragma unroll
    for (int ds = 0; ds < 4; ++ds)
        #pragma unroll
        for (int i = 0; i < 8; ++i) o[ds][i] = 0.f;

    const int vkey = tid >> 1;          // 0..127
    const int vseg = (tid & 1) * 32;    // 0 or 32
    const unsigned ksBase = (unsigned)(size_t)&Ks[0][0];
    const unsigned ksRowB = (64 + LDS_PAD) * 2;   // bytes per Ks row

    for (int kb = 0; kb < SEQ; kb += 128) {
        // prefetch next key block toward this WGP (global_prefetch_b8)
        if (kb + 128 < SEQ) {
            const size_t prow = bhoff + (size_t)(kb + 128 + (tid & 127)) * D_HEAD;
            __builtin_prefetch(K + prow, 0, 1);
            __builtin_prefetch(V + prow, 0, 1);
        }

        __syncthreads();   // previous iteration's Ks/Vt reads are complete

        // ---- async-stage K block to LDS: 4 x b128 per thread ----
        #pragma unroll
        for (int j = 0; j < 4; ++j) {
            const int id  = tid * 4 + j;    // 0..1023 chunks of 8 bf16
            const int key = id >> 3;
            const int seg = id & 7;
            async_copy_b128(ksBase + (unsigned)(key * ksRowB + seg * 16),
                            K + bhoff + (size_t)(kb + key) * D_HEAD + seg * 8);
        }

        // ---- stage V^T cooperatively: Vt[d][key] ----
        {
            const unsigned short* vr =
                V + bhoff + (size_t)(kb + vkey) * D_HEAD + vseg;
            #pragma unroll
            for (int i = 0; i < 32; ++i) Vt[vseg + i][vkey] = vr[i];
        }

        wait_async0();     // this wave's K copies landed in LDS
        __syncthreads();   // all waves' Ks/Vt staged

        // ---- S = Q K^T, 8 key subtiles of 16; Ks rows are B-columns ----
        v8f s[8];
        #pragma unroll
        for (int kn = 0; kn < 8; ++kn) {
            #pragma unroll
            for (int i = 0; i < 8; ++i) s[kn][i] = 0.f;
            const unsigned short* kr = &Ks[kn * 16 + lm][0];
            #pragma unroll
            for (int kc = 0; kc < 2; ++kc) {
                FragU kf;
                kf.q[0] = *(const uint4*)(kr + kc * 32 + 8 * half);
                kf.q[1] = *(const uint4*)(kr + kc * 32 + 16 + 8 * half);
                s[kn] = __builtin_amdgcn_wmma_f32_16x16x32_bf16(
                    false, qf[kc].v, false, kf.v, (short)0, s[kn], false, false);
            }
        }

        // ---- online softmax (row M = r + 8*half; reduce across 16 lanes) ----
        #pragma unroll
        for (int r = 0; r < 8; ++r) {
            float mx = s[0][r];
            #pragma unroll
            for (int kn = 1; kn < 8; ++kn) mx = fmaxf(mx, s[kn][r]);
            #pragma unroll
            for (int off = 8; off > 0; off >>= 1)
                mx = fmaxf(mx, __shfl_xor(mx, off, 32));
            const float mnew  = fmaxf(mstate[r], mx);
            const float alpha = __expf(mstate[r] - mnew);
            mstate[r] = mnew;
            float rs = 0.f;
            #pragma unroll
            for (int kn = 0; kn < 8; ++kn) {
                s[kn][r] = __expf(s[kn][r] - mnew);
                rs += s[kn][r];
            }
            #pragma unroll
            for (int off = 8; off > 0; off >>= 1)
                rs += __shfl_xor(rs, off, 32);
            lstate[r] = lstate[r] * alpha + rs;
            #pragma unroll
            for (int ds = 0; ds < 4; ++ds) o[ds][r] *= alpha;
        }

        // ---- P -> LDS (C layout -> memory) so we can reload as A-fragments ----
        #pragma unroll
        for (int kn = 0; kn < 8; ++kn)
            #pragma unroll
            for (int r = 0; r < 8; ++r)
                Pbuf[wave][r + 8 * half][kn * 16 + lm] = f32_to_bf16(s[kn][r]);

        // ---- O += P @ V  (K dim = 128 keys -> 4 chunks; 4 d-subtiles) ----
        #pragma unroll
        for (int kc = 0; kc < 4; ++kc) {
            FragU pf;
            const unsigned short* pp = &Pbuf[wave][lm][kc * 32];
            pf.q[0] = *(const uint4*)(pp + 8 * half);
            pf.q[1] = *(const uint4*)(pp + 16 + 8 * half);
            #pragma unroll
            for (int ds = 0; ds < 4; ++ds) {
                FragU vf;
                const unsigned short* vp = &Vt[ds * 16 + lm][kc * 32];
                vf.q[0] = *(const uint4*)(vp + 8 * half);
                vf.q[1] = *(const uint4*)(vp + 16 + 8 * half);
                o[ds] = __builtin_amdgcn_wmma_f32_16x16x32_bf16(
                    false, pf.v, false, vf.v, (short)0, o[ds], false, false);
            }
        }
    }

    // ---- normalize and write context, merged-head layout [B][SEQ][D_MODEL] ----
    const int b = bh >> 4, h = bh & 15;
    #pragma unroll
    for (int r = 0; r < 8; ++r) {
        const float inv = 1.f / lstate[r];
        const int qrow = qt * 128 + wave * 16 + r + 8 * half;
        #pragma unroll
        for (int ds = 0; ds < 4; ++ds) {
            const int col = h * 64 + ds * 16 + lm;
            Ctx[((size_t)(b * SEQ + qrow)) * D_MODEL + col] =
                f32_to_bf16(o[ds][r] * inv);
        }
    }
}

// ---------------------------------------------------------------------------
extern "C" void kernel_launch(void* const* d_in, const int* in_sizes, int n_in,
                              void* d_out, int out_size, void* d_ws, size_t ws_size,
                              hipStream_t stream)
{
    (void)in_sizes; (void)n_in; (void)out_size; (void)ws_size;

    const float* q  = (const float*)d_in[0];
    const float* k  = (const float*)d_in[1];
    const float* v  = (const float*)d_in[2];
    const float* Wq = (const float*)d_in[3];
    const float* bq = (const float*)d_in[4];
    const float* Wk = (const float*)d_in[5];
    const float* bk = (const float*)d_in[6];
    const float* Wv = (const float*)d_in[7];
    const float* bv = (const float*)d_in[8];
    const float* Wo = (const float*)d_in[9];
    const float* bo = (const float*)d_in[10];
    float* out = (float*)d_out;

    const size_t elems = (size_t)MROWS * D_MODEL;  // 4M bf16 each
    unsigned short* Qh  = (unsigned short*)d_ws;
    unsigned short* Kh  = Qh + elems;
    unsigned short* Vh  = Kh + elems;
    unsigned short* Ctx = Vh + elems;

    dim3 gGemm(D_MODEL / 128, MROWS / 128);        // (8, 32)
    // Q projection carries the 1/sqrt(d_head) softmax scale.
    gemm_wmma_kernel<float, 0><<<gGemm, 256, 0, stream>>>(q, Wq, bq, Qh, MROWS, D_MODEL, D_MODEL, 0.125f);
    gemm_wmma_kernel<float, 0><<<gGemm, 256, 0, stream>>>(k, Wk, bk, Kh, MROWS, D_MODEL, D_MODEL, 1.0f);
    gemm_wmma_kernel<float, 0><<<gGemm, 256, 0, stream>>>(v, Wv, bv, Vh, MROWS, D_MODEL, D_MODEL, 1.0f);

    dim3 gAttn(SEQ / 128, BATCH * NHEAD);          // (16, 32)
    flash_attn_kernel<<<gAttn, 256, 0, stream>>>(Qh, Kh, Vh, Ctx);

    gemm_wmma_kernel<unsigned short, 2><<<gGemm, 256, 0, stream>>>(Ctx, Wo, bo, out, MROWS, D_MODEL, D_MODEL, 1.0f);
}
